// WindowAttention_63513976373835
// MI455X (gfx1250) — compile-verified, matrix operand order checked by reading, not run on hardware
//
#include <hip/hip_runtime.h>
#include <hip/hip_bf16.h>

typedef __attribute__((ext_vector_type(16))) _Float16 v16h;
typedef __attribute__((ext_vector_type(8)))  _Float16 v8h;
typedef __attribute__((ext_vector_type(4)))  _Float16 v4h;
typedef __attribute__((ext_vector_type(8)))  float    v8f;

// ---------------------------------------------------------------------------
// Fragment builder for v_wmma_f32_16x16x32_f16 operands (A or B^T), wave32.
// Operand tile is row-major [16][>=32] f16 in LDS; 'tile' points at row 0 of
// the 16-row strip, 'stride' in halves. Per CDNA5 ISA 16-bit A layout:
//   lane<16 : row=lane,    K = {0..7}   (elems 0-7), {16..23} (elems 8-15)
//   lane>=16: row=lane-16, K = {8..15}  (elems 0-7), {24..31} (elems 8-15)
// All byte offsets are multiples of 16 -> ds_load_b128.
// ---------------------------------------------------------------------------
__device__ __forceinline__ v16h make_frag(const _Float16* tile, int stride) {
    const int lane = threadIdx.x & 31;
    const int rr   = lane & 15;
    const int kb   = (lane & 16) ? 8 : 0;
    const _Float16* p = tile + rr * stride + kb;
    v8h lo = *(const v8h*)(p);        // K = kb .. kb+7
    v8h hi = *(const v8h*)(p + 16);   // K = kb+16 .. kb+23
    v16h r;
#pragma unroll
    for (int i = 0; i < 8; ++i) { r[i] = lo[i]; r[i + 8] = hi[i]; }
    return r;
}

// ---------------------------------------------------------------------------
// Generic GEMM: out[M,N] = A[M,K] @ W[N,K]^T + bias[N]
// Block: 256 threads (8 waves). Block tile 128(M) x 64(N); each wave owns a
// 2x2 grid of 16x16 WMMA tiles (wy = wid&3 -> 32-row group, wx = wid>>2 ->
// 32-col group). 4 WMMAs per K-step from 4 fragment builds (A/B reuse).
// Requires M%128==0, N%64==0, K%32==0 (true for all uses here).
// ---------------------------------------------------------------------------
template <typename AT, typename OT>
__global__ void gemm_wmma_kernel(const AT* __restrict__ A,
                                 const float* __restrict__ W,
                                 const float* __restrict__ bias,
                                 OT* __restrict__ out,
                                 int M, int N, int K) {
    __shared__ __attribute__((aligned(16))) _Float16 As[128 * 32];   // 8 KB
    __shared__ __attribute__((aligned(16))) _Float16 Ws[64 * 32];    // 4 KB

    const int tid = threadIdx.x;
    const int wid = tid >> 5;
    const int wy  = wid & 3;       // 32-row group 0..3
    const int wx  = wid >> 2;      // 32-col group 0..1
    const int m_block = blockIdx.x * 128;
    const int n_block = blockIdx.y * 64;

    v8f acc[2][2] = {{{}, {}}, {{}, {}}};

    for (int k0 = 0; k0 < K; k0 += 32) {
        // ---- stage A tile (128 x 32) as f16 -----------------------------
        if constexpr (sizeof(AT) == 4) {
            for (int t = tid; t < (128 * 32) / 4; t += 256) {
                int r = t >> 3, c = (t & 7) << 2;
                const float4 f =
                    *(const float4*)&A[(size_t)(m_block + r) * K + k0 + c];
                v4h hv = {(_Float16)f.x, (_Float16)f.y,
                          (_Float16)f.z, (_Float16)f.w};
                *(v4h*)&As[r * 32 + c] = hv;
            }
        } else {
            for (int t = tid; t < (128 * 32) / 8; t += 256) {
                int r = t >> 2, c = (t & 3) << 3;
                *(v8h*)&As[r * 32 + c] =
                    *(const v8h*)&A[(size_t)(m_block + r) * K + k0 + c];
            }
        }
        // ---- stage W tile (64 out-rows x 32 K) as f16 (B^T row-major) ---
        for (int t = tid; t < (64 * 32) / 4; t += 256) {
            int r = t >> 3, c = (t & 7) << 2;
            const float4 f =
                *(const float4*)&W[(size_t)(n_block + r) * K + k0 + c];
            v4h hv = {(_Float16)f.x, (_Float16)f.y,
                      (_Float16)f.z, (_Float16)f.w};
            *(v4h*)&Ws[r * 32 + c] = hv;
        }
        // Prefetch next K-slab of A toward L2 (global_prefetch_b8)
        if (k0 + 32 < K)
            __builtin_prefetch(
                &A[(size_t)(m_block + (tid & 127)) * K + k0 + 32], 0, 1);
        __syncthreads();

        v16h af[2], bf[2];
        af[0] = make_frag(&As[(32 * wy) * 32], 32);
        af[1] = make_frag(&As[(32 * wy + 16) * 32], 32);
        bf[0] = make_frag(&Ws[(32 * wx) * 32], 32);
        bf[1] = make_frag(&Ws[(32 * wx + 16) * 32], 32);
#pragma unroll
        for (int mi = 0; mi < 2; ++mi)
#pragma unroll
            for (int ni = 0; ni < 2; ++ni)
                acc[mi][ni] = __builtin_amdgcn_wmma_f32_16x16x32_f16(
                    false, af[mi], false, bf[ni], (short)0, acc[mi][ni],
                    false, false);

        __syncthreads();
    }

    // Epilogue: C/D layout -> lane<16: (M=r, N=lane); lane>=16: (M=r+8, N=lane-16)
    const int lane = tid & 31;
#pragma unroll
    for (int ni = 0; ni < 2; ++ni) {
        const int ncol = n_block + 32 * wx + 16 * ni + (lane & 15);
        const float bv = bias[ncol];
#pragma unroll
        for (int mi = 0; mi < 2; ++mi) {
            const int mbase =
                m_block + 32 * wy + 16 * mi + ((lane & 16) ? 8 : 0);
#pragma unroll
            for (int r = 0; r < 8; ++r)
                out[(size_t)(mbase + r) * N + ncol] =
                    (OT)(acc[mi][ni][r] + bv);
        }
    }
}

// ---------------------------------------------------------------------------
// Attention core: one block = one (window b, head h). 4 waves x 32 = 128 thr.
// qkv f16 layout: row t=b*49+i, cols [q(0..255)|k(256..511)|v(512..767)],
// head h occupies 32 contiguous cols h*32.
// ---------------------------------------------------------------------------
__global__ void attn_kernel(const _Float16* __restrict__ qkv,
                            const float* __restrict__ wmask,     // [64,49,49]
                            const float* __restrict__ fg,        // [B_,49,49]
                            const float* __restrict__ bg,        // [B_,49,49]
                            const float* __restrict__ table,     // [169,8]
                            _Float16* __restrict__ y) {          // [B_*49,256]
    __shared__ __attribute__((aligned(16))) _Float16 qh[64 * 32];
    __shared__ __attribute__((aligned(16))) _Float16 kh[64 * 32];
    __shared__ __attribute__((aligned(16))) _Float16 vT[32 * 64];
    __shared__ __attribute__((aligned(16))) _Float16 ps[4][16 * 64];

    const int bh = blockIdx.x;
    const int b  = bh >> 3;
    const int h  = bh & 7;
    const int tid = threadIdx.x;

    const size_t rowbase = (size_t)b * 49 * 768;
    const int qoff = h * 32, koff = 256 + h * 32, voff = 512 + h * 32;
    const float scale = 0.17677669529663687f;   // 1/sqrt(32)

    // Stage q (scaled), k as [64 rows x 32], v transposed as [32 x 64]
    for (int idx = tid; idx < 64 * 32; idx += 128) {
        int i = idx >> 5, d = idx & 31;
        float qv = 0.f, kv = 0.f;
        if (i < 49) {
            size_t base = rowbase + (size_t)i * 768;
            qv = (float)qkv[base + qoff + d] * scale;
            kv = (float)qkv[base + koff + d];
        }
        qh[idx] = (_Float16)qv;
        kh[idx] = (_Float16)kv;
    }
    for (int idx = tid; idx < 32 * 64; idx += 128) {
        int d = idx >> 6, i = idx & 63;
        float vv = (i < 49) ? (float)qkv[rowbase + (size_t)i * 768 + voff + d]
                            : 0.f;
        vT[idx] = (_Float16)vv;
    }
    __syncthreads();

    const int w    = tid >> 5;
    const int lane = tid & 31;
    const int m0   = 16 * w;
    const int nloc = lane & 15;
    const int mrow = m0 + ((lane & 16) ? 8 : 0);   // + r gives global row
    const int wdw  = b & 63;                        // b % nW

    // S = q @ k^T  (4 column tiles of 16)
    v16h aq = make_frag(&qh[m0 * 32], 32);
    float S[4][8];
#pragma unroll
    for (int j = 0; j < 4; ++j) {
        v16h bk = make_frag(&kh[(16 * j) * 32], 32);
        v8f acc = {};
        acc = __builtin_amdgcn_wmma_f32_16x16x32_f16(
            false, aq, false, bk, (short)0, acc, false, false);
#pragma unroll
        for (int r = 0; r < 8; ++r) S[j][r] = acc[r];
    }

    // Add rel-pos bias + window mask (invalid elements -> -inf surrogate)
    const float NEG = -1e30f;
#pragma unroll
    for (int j = 0; j < 4; ++j) {
        int n = 16 * j + nloc;
#pragma unroll
        for (int r = 0; r < 8; ++r) {
            int m = mrow + r;
            if (m < 49 && n < 49) {
                int rel = ((m / 7 - n / 7) + 6) * 13 + ((m % 7 - n % 7) + 6);
                S[j][r] += table[rel * 8 + h] +
                           wmask[((size_t)wdw * 49 + m) * 49 + n];
            } else {
                S[j][r] = NEG;
            }
        }
    }

    // Three softmaxes (plain, +fg, +bg), combined: P = Pa + Pfg - Pbg.
    // Saliency masks loaded on demand (each element read exactly once) to
    // keep VGPR pressure low. Each score row lives in 16 lanes x 4 j-regs.
    float P[4][8];
#pragma unroll
    for (int j = 0; j < 4; ++j)
#pragma unroll
        for (int r = 0; r < 8; ++r) P[j][r] = 0.f;

#pragma unroll
    for (int v = 0; v < 3; ++v) {
        const float sign = (v == 2) ? -1.f : 1.f;
        const float* smask = (v == 1) ? fg : bg;
#pragma unroll
        for (int r = 0; r < 8; ++r) {
            const int m = mrow + r;
            float vals[4];
#pragma unroll
            for (int j = 0; j < 4; ++j) {
                const int n = 16 * j + nloc;
                float e = 0.f;
                if (v > 0 && m < 49 && n < 49)
                    e = smask[((size_t)b * 49 + m) * 49 + n];
                vals[j] = S[j][r] + e;
            }
            float mx = fmaxf(fmaxf(vals[0], vals[1]), fmaxf(vals[2], vals[3]));
#pragma unroll
            for (int s = 1; s < 16; s <<= 1)
                mx = fmaxf(mx, __shfl_xor(mx, s, 32));
            if (mx < -1e29f) mx = 0.f;     // fully-masked (padding) row
            float es[4], sum = 0.f;
#pragma unroll
            for (int j = 0; j < 4; ++j) {
                es[j] = __expf(vals[j] - mx);
                sum += es[j];
            }
#pragma unroll
            for (int s = 1; s < 16; s <<= 1)
                sum += __shfl_xor(sum, s, 32);
            const float inv = sign / fmaxf(sum, 1e-20f);
#pragma unroll
            for (int j = 0; j < 4; ++j) P[j][r] += es[j] * inv;
        }
    }

    // C-layout -> A-layout via wave-private LDS strip [16][64]
    _Float16* pw = ps[w];
#pragma unroll
    for (int j = 0; j < 4; ++j) {
        const int n = 16 * j + nloc;
#pragma unroll
        for (int r = 0; r < 8; ++r) {
            const int ml = (mrow - m0) + r;
            pw[ml * 64 + n] = (_Float16)P[j][r];
        }
    }
    // (wave-private region: in-wave DS ordering + compiler waitcnts suffice)

    // out_strip[16x32] = P[16x64] @ V[64x32]   (K=64 in two 32-chunks)
    v16h ap0 = make_frag(pw, 64);
    v16h ap1 = make_frag(pw + 32, 64);
#pragma unroll
    for (int t = 0; t < 2; ++t) {
        v16h bv0 = make_frag(&vT[(16 * t) * 64], 64);
        v16h bv1 = make_frag(&vT[(16 * t) * 64 + 32], 64);
        v8f acc = {};
        acc = __builtin_amdgcn_wmma_f32_16x16x32_f16(
            false, ap0, false, bv0, (short)0, acc, false, false);
        acc = __builtin_amdgcn_wmma_f32_16x16x32_f16(
            false, ap1, false, bv1, (short)0, acc, false, false);
        const int d = 16 * t + nloc;
#pragma unroll
        for (int r = 0; r < 8; ++r) {
            const int m = mrow + r;
            if (m < 49)
                y[((size_t)b * 49 + m) * 256 + h * 32 + d] = (_Float16)acc[r];
        }
    }
}

// ---------------------------------------------------------------------------
// Host launch. Inputs (setup_inputs order):
// 0:x 1:mask 2:sal_fg 3:sal_bg 4:qkv_w 5:qkv_b 6:proj_w 7:proj_b 8:rel_table
// ---------------------------------------------------------------------------
extern "C" void kernel_launch(void* const* d_in, const int* in_sizes, int n_in,
                              void* d_out, int out_size, void* d_ws, size_t ws_size,
                              hipStream_t stream) {
    const float* x      = (const float*)d_in[0];
    const float* wmask  = (const float*)d_in[1];
    const float* fg     = (const float*)d_in[2];
    const float* bg     = (const float*)d_in[3];
    const float* qkv_w  = (const float*)d_in[4];
    const float* qkv_b  = (const float*)d_in[5];
    const float* proj_w = (const float*)d_in[6];
    const float* proj_b = (const float*)d_in[7];
    const float* table  = (const float*)d_in[8];
    float* out = (float*)d_out;

    const int B_ = 2048, Ntok = 49, C = 256;
    const int M = B_ * Ntok;              // 100352 = 128 * 784

    _Float16* qkv = (_Float16*)d_ws;                       // [M, 768] f16
    _Float16* y   = qkv + (size_t)M * 3 * C;               // [M, 256] f16

    // 1) QKV projection (f32 in -> f16 out)
    gemm_wmma_kernel<float, _Float16>
        <<<dim3(M / 128, (3 * C) / 64), dim3(256), 0, stream>>>(
            x, qkv_w, qkv_b, qkv, M, 3 * C, C);

    // 2) Windowed attention (per window-head block)
    attn_kernel<<<dim3(B_ * 8), dim3(128), 0, stream>>>(
        qkv, wmask, fg, bg, table, y);

    // 3) Output projection (f16 in -> f32 out)
    gemm_wmma_kernel<_Float16, float>
        <<<dim3(M / 128, C / 64), dim3(256), 0, stream>>>(
            y, proj_w, proj_b, out, M, C, C);
}
